// mLSTM_24592982737311
// MI455X (gfx1250) — compile-verified, matrix operand order checked by reading, not run on hardware
//
#include <hip/hip_runtime.h>
#include <hip/hip_bf16.h>
#include <cstdint>

// ---------------------------------------------------------------------------
// 2-layer LSTM scan on gfx1250 (MI455X): bf16 WMMA + fp32 accum, async-to-LDS
// double-buffered staging, fused LSTM cell epilogue.
// ---------------------------------------------------------------------------

typedef __attribute__((ext_vector_type(16))) __bf16 v16bf;
typedef __attribute__((ext_vector_type(8)))  float  v8f;

#define EP   512   // scan steps
#define TT   256   // batch rows per step
#define NIN  512   // input dim
#define HH   1024  // hidden dim

__device__ __forceinline__ float sigf(float x)     { return 1.0f / (1.0f + __expf(-x)); }
__device__ __forceinline__ float tanhfast(float x) { return 1.0f - 2.0f / (__expf(2.0f * x) + 1.0f); }

// gfx1250 async DMA global->LDS (ASYNCcnt-tracked, no VGPR round trip).
// ldsoff is a raw byte offset into the workgroup LDS allocation.
__device__ __forceinline__ void async_ld_b128(unsigned ldsoff, const void* g) {
  asm volatile("global_load_async_to_lds_b128 %0, %1, off"
               :: "v"(ldsoff), "v"((unsigned long long)(uintptr_t)g)
               : "memory");
}
__device__ __forceinline__ void wait_async0() {
  asm volatile("s_wait_asynccnt 0x0" ::: "memory");
}

// ---------------------------------------------------------------------------
// One-time prep: weights -> bf16, combined biases, S = sum(W_tl), zero states.
// ---------------------------------------------------------------------------
__global__ void prep_kernel(const float* __restrict__ Wih1, const float* __restrict__ Whh1,
                            const float* __restrict__ Wih2, const float* __restrict__ Whh2,
                            const float* __restrict__ Whl,
                            const float* __restrict__ bih1, const float* __restrict__ bhh1,
                            const float* __restrict__ bih2, const float* __restrict__ bhh2,
                            const float* __restrict__ wtl,
                            __bf16* Wih1b, __bf16* Whh1b, __bf16* Wih2b, __bf16* Whh2b,
                            __bf16* Whlb, float* bsum1, float* bsum2, float* Ssum,
                            __bf16* h1a, __bf16* h1b, __bf16* h2a, __bf16* h2b,
                            float* c1, float* c2, float* V) {
  const size_t tid    = (size_t)blockIdx.x * blockDim.x + threadIdx.x;
  const size_t stride = (size_t)gridDim.x * blockDim.x;
  for (size_t i = tid; i < (size_t)4 * HH * NIN; i += stride) Wih1b[i] = (__bf16)Wih1[i];
  for (size_t i = tid; i < (size_t)4 * HH * HH; i += stride) {
    Whh1b[i] = (__bf16)Whh1[i];
    Wih2b[i] = (__bf16)Wih2[i];
    Whh2b[i] = (__bf16)Whh2[i];
  }
  for (size_t i = tid; i < (size_t)NIN * HH; i += stride) Whlb[i] = (__bf16)Whl[i];
  for (size_t i = tid; i < (size_t)4 * HH; i += stride) {
    bsum1[i] = bih1[i] + bhh1[i];
    bsum2[i] = bih2[i] + bhh2[i];
  }
  for (size_t i = tid; i < (size_t)TT * HH; i += stride) {
    h1a[i] = (__bf16)0.0f; h1b[i] = (__bf16)0.0f;
    h2a[i] = (__bf16)0.0f; h2b[i] = (__bf16)0.0f;
    c1[i] = 0.0f; c2[i] = 0.0f;
  }
  for (size_t i = tid; i < (size_t)EP * HH; i += stride) V[i] = 0.0f;
  if (tid == 0) {
    float s = 0.0f;
    for (int t = 0; t < TT; ++t) s += wtl[t];
    *Ssum = s;
  }
}

// ---------------------------------------------------------------------------
// Fused gates-GEMM + LSTM cell update.
//   gates(256 x 4096) = [A1 | A2] @ [Wih | Whh]^T, then i,f,g,o -> c,h.
// Block: 256 thr = 8 waves; tile 64 rows x (4 gates x 32 cols).
// K loop: 64-wide chunks, double-buffered LDS, async-to-LDS prefetch of
// chunk k+1 overlapped with WMMA compute of chunk k (8 WMMAs / chunk / wave).
// A1F32: first K-phase reads fp32 (raw inputs) and converts while staging.
// VRED : fuse v_e[j] += wtl[t]*h[t,j]  (layer 2 only).
// ---------------------------------------------------------------------------
template <bool A1F32, bool VRED>
__global__ __launch_bounds__(256) void lstm_layer_kernel(
    const float* __restrict__ A1f, const __bf16* __restrict__ A1b, int K1,
    const __bf16* __restrict__ A2b,             // TT x HH (prev h, bf16)
    const __bf16* __restrict__ Wih,             // 4HH x K1
    const __bf16* __restrict__ Whh,             // 4HH x HH
    const float* __restrict__ bsum,             // 4HH
    float* __restrict__ Cst,                    // TT x HH cell state (fp32, in/out)
    __bf16* __restrict__ Hout,                  // TT x HH new h (bf16)
    const float* __restrict__ wtl,              // TT
    float* __restrict__ Vrow)                   // HH (atomic accum), layer 2 only
{
  const int tid  = threadIdx.x;
  const int lane = tid & 31;
  const int wave = tid >> 5;
  const int wm   = wave & 3;
  const int wj   = wave >> 2;
  const int m0   = blockIdx.x * 64;
  const int j0   = blockIdx.y * 32;

  // Single LDS allocation => base byte offset 0 for async addressing.
  // Element layout: A0 [0,4096) A1 [4096,8192) B0 [8192,16384) B1 [16384,24576)
  __shared__ __align__(32) __bf16 smem[24576];   // 48 KB of the 320 KB/WGP

  const int KT = K1 + HH;
  const int NC = KT >> 6;                        // 64-wide K chunks

  auto stage = [&](int kb, int buf) {
    // ---- B tile: 128 rows (gate g, col c -> W row g*HH+j0+c) x 64 k ----
    const unsigned bbyte = 16384u + (unsigned)buf * 16384u;
#pragma unroll
    for (int uu = 0; uu < 4; ++uu) {
      const int u  = tid + uu * 256;
      const int rb = u >> 3;                     // 0..127
      const int q  = u & 7;                      // 16B unit within row
      const int g  = rb >> 5, c = rb & 31;
      const __bf16* src = (kb < K1)
          ? Wih + (size_t)(g * HH + j0 + c) * K1 + kb + q * 8
          : Whh + (size_t)(g * HH + j0 + c) * HH + (kb - K1) + q * 8;
      async_ld_b128(bbyte + (unsigned)(rb * 128 + q * 16), src);
    }
    // ---- A tile: 64 rows x 64 k ----
    if (A1F32 && kb < K1) {
      // fp32 inputs: load + convert + ds_store (sync path, DScnt-ordered)
      const int row = tid >> 2, q = tid & 3;
      const float4* s4 = (const float4*)(A1f + (size_t)(m0 + row) * K1 + kb + q * 16);
      v16bf t;
#pragma unroll
      for (int qq = 0; qq < 4; ++qq) {
        const float4 f = s4[qq];
        t[4 * qq + 0] = (__bf16)f.x; t[4 * qq + 1] = (__bf16)f.y;
        t[4 * qq + 2] = (__bf16)f.z; t[4 * qq + 3] = (__bf16)f.w;
      }
      *(v16bf*)&smem[buf * 4096 + row * 64 + q * 16] = t;
    } else {
      const unsigned abyte = (unsigned)buf * 8192u;
#pragma unroll
      for (int uu = 0; uu < 2; ++uu) {
        const int u   = tid + uu * 256;
        const int row = u >> 3, q = u & 7;
        const __bf16* src = (kb < K1)
            ? A1b + (size_t)(m0 + row) * K1 + kb + q * 8
            : A2b + (size_t)(m0 + row) * HH + (kb - K1) + q * 8;
        async_ld_b128(abyte + (unsigned)(row * 128 + q * 16), src);
      }
    }
  };

  v8f zero = {};
  v8f acc[4];
#pragma unroll
  for (int g = 0; g < 4; ++g) acc[g] = zero;

  stage(0, 0);

  const int kx   = (lane >> 4) * 16;
  const int arow = wm * 16 + (lane & 15);

  for (int kc = 0; kc < NC; ++kc) {
    const int cur = kc & 1;
    wait_async0();          // my async stage of chunk kc complete
    __syncthreads();        // visible to all waves; buffer cur^1 free to reuse
    if (kc + 1 < NC) stage((kc + 1) * 64, cur ^ 1);   // overlap with compute
#pragma unroll
    for (int ks = 0; ks < 2; ++ks) {
      v16bf a = *(const v16bf*)&smem[cur * 4096 + arow * 64 + ks * 32 + kx];
#pragma unroll
      for (int g = 0; g < 4; ++g) {
        const int rb = g * 32 + wj * 16 + (lane & 15);
        v16bf b = *(const v16bf*)&smem[8192 + cur * 8192 + rb * 64 + ks * 32 + kx];
        acc[g] = __builtin_amdgcn_wmma_f32_16x16x32_bf16(false, a, false, b, (short)0,
                                                         acc[g], false, false);
      }
    }
  }

  // ---- epilogue: bias + lane-local LSTM cell update ----
  const int jcol = j0 + wj * 16 + (lane & 15);
  const float bi = bsum[0 * HH + jcol];
  const float bf = bsum[1 * HH + jcol];
  const float bg = bsum[2 * HH + jcol];
  const float bo = bsum[3 * HH + jcol];
  float vsum = 0.0f;
#pragma unroll
  for (int v = 0; v < 8; ++v) {
    const int r = m0 + wm * 16 + v + 8 * (lane >> 4);   // C-layout: M = v (+8 hi half)
    const size_t idx = (size_t)r * HH + jcol;
    const float iv = sigf(acc[0][v] + bi);
    const float fv = sigf(acc[1][v] + bf);
    const float gv = tanhfast(acc[2][v] + bg);
    const float ov = sigf(acc[3][v] + bo);
    const float cn = fv * Cst[idx] + iv * gv;
    const float hn = ov * tanhfast(cn);
    Cst[idx]  = cn;
    Hout[idx] = (__bf16)hn;
    if (VRED) vsum += wtl[r] * hn;
  }
  if (VRED) atomicAdd(&Vrow[jcol], vsum);
}

// ---------------------------------------------------------------------------
// V (fp32) -> bf16
// ---------------------------------------------------------------------------
__global__ void cvt_kernel(const float* __restrict__ src, __bf16* __restrict__ dst, int n) {
  int i = blockIdx.x * blockDim.x + threadIdx.x;
  const int stride = gridDim.x * blockDim.x;
  for (; i < n; i += stride) dst[i] = (__bf16)src[i];
}

// ---------------------------------------------------------------------------
// Final: out(EP x NIN) = Vb(EP x HH) @ Whlb^T + b_hl[n]*S + b_tl   (runs once)
// ---------------------------------------------------------------------------
__global__ __launch_bounds__(256) void out_gemm_kernel(
    const __bf16* __restrict__ Vb, const __bf16* __restrict__ Wb,
    const float* __restrict__ bhl, const float* __restrict__ Ssum,
    const float* __restrict__ btl, float* __restrict__ out) {
  const int tid  = threadIdx.x;
  const int lane = tid & 31;
  const int wave = tid >> 5;
  const int wm   = wave & 3;
  const int wj   = wave >> 2;
  const int m0   = blockIdx.x * 64;   // over EP
  const int n0   = blockIdx.y * 64;   // over NIN

  __shared__ __align__(32) __bf16 ldsA[64 * 32];
  __shared__ __align__(32) __bf16 ldsB[64 * 32];

  v8f zero = {};
  v8f acc[2] = {zero, zero};

  for (int k = 0; k < HH; k += 32) {
    __syncthreads();
    if (tid < 128) {
      const int r = tid >> 1, kh = tid & 1;
      *(v16bf*)&ldsA[r * 32 + kh * 16] =
          *(const v16bf*)(Vb + (size_t)(m0 + r) * HH + k + kh * 16);
    } else {
      const int t2 = tid - 128;
      const int r = t2 >> 1, kh = t2 & 1;
      *(v16bf*)&ldsB[r * 32 + kh * 16] =
          *(const v16bf*)(Wb + (size_t)(n0 + r) * HH + k + kh * 16);
    }
    __syncthreads();
    const int kx = (lane >> 4) * 16;
    v16bf a = *(const v16bf*)&ldsA[(wm * 16 + (lane & 15)) * 32 + kx];
#pragma unroll
    for (int jt = 0; jt < 2; ++jt) {
      v16bf b = *(const v16bf*)&ldsB[(wj * 32 + jt * 16 + (lane & 15)) * 32 + kx];
      acc[jt] = __builtin_amdgcn_wmma_f32_16x16x32_bf16(false, a, false, b, (short)0,
                                                        acc[jt], false, false);
    }
  }
  const float S = *Ssum, bt = *btl;
#pragma unroll
  for (int jt = 0; jt < 2; ++jt) {
    const int n = n0 + wj * 32 + jt * 16 + (lane & 15);
    const float bb = bhl[n] * S + bt;
#pragma unroll
    for (int v = 0; v < 8; ++v) {
      const int r = m0 + wm * 16 + v + 8 * (lane >> 4);
      out[(size_t)r * NIN + n] = acc[jt][v] + bb;
    }
  }
}

// ---------------------------------------------------------------------------
extern "C" void kernel_launch(void* const* d_in, const int* in_sizes, int n_in,
                              void* d_out, int out_size, void* d_ws, size_t ws_size,
                              hipStream_t stream) {
  (void)in_sizes; (void)n_in; (void)out_size; (void)ws_size;

  const float* inputs = (const float*)d_in[0];
  const float* W_ih1  = (const float*)d_in[1];
  const float* b_ih1  = (const float*)d_in[2];
  const float* W_hh1  = (const float*)d_in[3];
  const float* b_hh1  = (const float*)d_in[4];
  const float* W_ih2  = (const float*)d_in[5];
  const float* b_ih2  = (const float*)d_in[6];
  const float* W_hh2  = (const float*)d_in[7];
  const float* b_hh2  = (const float*)d_in[8];
  const float* W_hl   = (const float*)d_in[9];
  const float* b_hl   = (const float*)d_in[10];
  const float* W_tl   = (const float*)d_in[11];
  const float* b_tl   = (const float*)d_in[12];

  // workspace carve-up (256B aligned slots)
  char*  ws  = (char*)d_ws;
  size_t off = 0;
  auto carve = [&](size_t bytes) -> char* {
    char* p = ws + off;
    off = (off + bytes + 255) & ~(size_t)255;
    return p;
  };
  __bf16* Wih1b = (__bf16*)carve((size_t)4 * HH * NIN * 2);
  __bf16* Whh1b = (__bf16*)carve((size_t)4 * HH * HH * 2);
  __bf16* Wih2b = (__bf16*)carve((size_t)4 * HH * HH * 2);
  __bf16* Whh2b = (__bf16*)carve((size_t)4 * HH * HH * 2);
  __bf16* Whlb  = (__bf16*)carve((size_t)NIN * HH * 2);
  float*  bsum1 = (float*)carve((size_t)4 * HH * 4);
  float*  bsum2 = (float*)carve((size_t)4 * HH * 4);
  float*  Ssum  = (float*)carve(256);
  __bf16* h1[2] = {(__bf16*)carve((size_t)TT * HH * 2), (__bf16*)carve((size_t)TT * HH * 2)};
  __bf16* h2[2] = {(__bf16*)carve((size_t)TT * HH * 2), (__bf16*)carve((size_t)TT * HH * 2)};
  float*  c1    = (float*)carve((size_t)TT * HH * 4);
  float*  c2    = (float*)carve((size_t)TT * HH * 4);
  float*  V     = (float*)carve((size_t)EP * HH * 4);
  __bf16* Vb    = (__bf16*)carve((size_t)EP * HH * 2);

  prep_kernel<<<1024, 256, 0, stream>>>(W_ih1, W_hh1, W_ih2, W_hh2, W_hl,
                                        b_ih1, b_hh1, b_ih2, b_hh2, W_tl,
                                        Wih1b, Whh1b, Wih2b, Whh2b, Whlb,
                                        bsum1, bsum2, Ssum,
                                        h1[0], h1[1], h2[0], h2[1], c1, c2, V);

  const dim3 grid(TT / 64, HH / 32);   // 4 x 32 = 128 blocks
  for (int e = 0; e < EP; ++e) {
    const float* x = inputs + (size_t)e * TT * NIN;
    __bf16* h1r = h1[(e + 1) & 1];
    __bf16* h1w = h1[e & 1];
    __bf16* h2r = h2[(e + 1) & 1];
    __bf16* h2w = h2[e & 1];
    lstm_layer_kernel<true, false><<<grid, 256, 0, stream>>>(
        x, nullptr, NIN, h1r, Wih1b, Whh1b, bsum1, c1, h1w, W_tl, nullptr);
    lstm_layer_kernel<false, true><<<grid, 256, 0, stream>>>(
        nullptr, h1w, HH, h2r, Wih2b, Whh2b, bsum2, c2, h2w, W_tl, V + (size_t)e * HH);
  }

  cvt_kernel<<<512, 256, 0, stream>>>(V, Vb, EP * HH);
  out_gemm_kernel<<<dim3(EP / 64, NIN / 64), 256, 0, stream>>>(
      Vb, Whlb, b_hl, Ssum, b_tl, (float*)d_out);
}
